// TaskAlignedAssigner_16741782520464
// MI455X (gfx1250) — matrix operand order repaired and checked by reading, not compiled, not run on hardware
//
#include <hip/hip_runtime.h>
#include <hip/hip_bf16.h>
#include <stdint.h>

#define NB 32
#define NA 8400
#define NGT 64
#define NCLS 80
#define TOPK 13

typedef __attribute__((ext_vector_type(16))) _Float16 v16h;
typedef __attribute__((ext_vector_type(8)))  float    v8f;

// align metric for a valid (in-gt) pair: score^1 * iou^6, iou clipped >= 0
__device__ __forceinline__ float metric_valid(
    float gx1, float gy1, float gx2, float gy2, float garea,
    float px1, float py1, float px2, float py2, float score)
{
    float ix1 = fmaxf(gx1, px1), iy1 = fmaxf(gy1, py1);
    float ix2 = fminf(gx2, px2), iy2 = fminf(gy2, py2);
    float iw = fmaxf(ix2 - ix1, 0.f), ih = fmaxf(iy2 - iy1, 0.f);
    float inter = iw * ih;
    float a2 = (px2 - px1) * (py2 - py1);
    float iou = inter / (garea + a2 - inter + 1e-7f);
    iou = fmaxf(iou, 0.f);
    float i2 = iou * iou;
    return score * (i2 * i2 * i2);
}

// ---------------------------------------------------------------------------
// Kernel A: one block per (b, n). Computes masked align metric over all 8400
// anchors into LDS, scatters the validity bit (anchor center strictly inside
// gt AND mask_gt>0) into valid_bits[b][a], then does TOPK iterative block
// argmax (ties -> lowest index, matching jax.lax.top_k) and scatters topk bits.
// ---------------------------------------------------------------------------
__global__ __launch_bounds__(256) void topk_kernel(
    const float* __restrict__ pd_scores, const float* __restrict__ pd_bboxes,
    const float* __restrict__ anc, const int* __restrict__ gt_labels,
    const float* __restrict__ gt_bboxes, const float* __restrict__ mask_gt,
    unsigned long long* __restrict__ valid_bits,
    unsigned long long* __restrict__ topk_bits)
{
    const int bn = blockIdx.x;
    const int b = bn >> 6;           // /64
    const int n = bn & 63;
    const int tid = threadIdx.x;

    __shared__ float met[NA];
    __shared__ float rv[256];
    __shared__ int   ri[256];

    const float gx1 = gt_bboxes[bn * 4 + 0];
    const float gy1 = gt_bboxes[bn * 4 + 1];
    const float gx2 = gt_bboxes[bn * 4 + 2];
    const float gy2 = gt_bboxes[bn * 4 + 3];
    const int   lab = gt_labels[bn];
    const float mgt = mask_gt[bn];
    const float garea = (gx2 - gx1) * (gy2 - gy1);
    const unsigned long long nb = 1ull << n;

    for (int a = tid; a < NA; a += 256) {
        float ax = anc[2 * a], ay = anc[2 * a + 1];
        bool ing = (ax - gx1 > 1e-9f) && (ay - gy1 > 1e-9f) &&
                   (gx2 - ax > 1e-9f) && (gy2 - ay > 1e-9f);
        bool valid = ing && (mgt > 0.f);
        float m = 0.f;
        if (valid) {
            const size_t ba = (size_t)b * NA + a;
            const float* pb = pd_bboxes + ba * 4;
            float score = pd_scores[ba * NCLS + lab];
            m = metric_valid(gx1, gy1, gx2, gy2, garea,
                             pb[0], pb[1], pb[2], pb[3], score);
            atomicOr(&valid_bits[ba], nb);
        }
        met[a] = m;
    }
    __syncthreads();

    for (int k = 0; k < TOPK; ++k) {
        float bv = -2.f;           // metrics are >= 0; removed slots are -1
        int   bi = 0x7fffffff;
        for (int a = tid; a < NA; a += 256) {
            float v = met[a];
            if (v > bv) { bv = v; bi = a; }   // strict > keeps lowest index
        }
        rv[tid] = bv; ri[tid] = bi;
        __syncthreads();
        for (int s = 128; s > 0; s >>= 1) {
            if (tid < s) {
                float ov = rv[tid + s]; int oi = ri[tid + s];
                if (ov > rv[tid] || (ov == rv[tid] && oi < ri[tid])) {
                    rv[tid] = ov; ri[tid] = oi;
                }
            }
            __syncthreads();
        }
        if (tid == 0) {
            int idx = ri[0];
            atomicOr(&topk_bits[(size_t)b * NA + idx], nb);
            met[idx] = -1.f;       // remove winner
        }
        __syncthreads();
    }
}

// ---------------------------------------------------------------------------
// Kernel B: one thread per (b, a). GT boxes for batch b are staged into LDS
// with CDNA5 async global->LDS loads (ASYNCcnt). Iterates only the set bits
// of the validity mask (sparse), does the multi-GT conflict resolution with
// 64-bit mask popcount/ctz, writes target_bboxes + fg_mask and the per-anchor
// (target_gt_idx, pos_align) consumed by the WMMA kernel.
// ---------------------------------------------------------------------------
__global__ __launch_bounds__(256) void assign_kernel(
    const float* __restrict__ pd_scores, const float* __restrict__ pd_bboxes,
    const int* __restrict__ gt_labels, const float* __restrict__ gt_bboxes,
    const unsigned long long* __restrict__ valid_bits,
    const unsigned long long* __restrict__ topk_bits,
    float* __restrict__ out_bboxes, float* __restrict__ out_fg,
    int* __restrict__ tgt_idx, float* __restrict__ pos_align_out)
{
    const int b = blockIdx.y;
    const int a = blockIdx.x * 256 + threadIdx.x;

    __shared__ float gtb[NGT * 4];   // 256 dwords
    __shared__ int   lab[NGT];

    // ---- CDNA5 async global -> LDS staging of gt boxes (ASYNCcnt) ----
    {
        const float* src = gt_bboxes + (size_t)b * NGT * 4 + threadIdx.x;
        uint32_t lds_off = (uint32_t)(uintptr_t)&gtb[threadIdx.x]; // low 32b of flat shared addr = LDS offset (ISA 10.2)
        asm volatile("global_load_async_to_lds_b32 %0, %1, off"
                     :: "v"(lds_off), "v"(src) : "memory");
        if (threadIdx.x < NGT) lab[threadIdx.x] = gt_labels[b * NGT + threadIdx.x];
        asm volatile("s_wait_asynccnt 0x0" ::: "memory");
    }
    __syncthreads();
    if (a >= NA) return;

    const size_t ba = (size_t)b * NA + a;
    const unsigned long long vb = valid_bits[ba];
    unsigned long long pos = vb & topk_bits[ba];

    float maxAll = 0.f, maxPos = 0.f;
    float px1 = 0.f, py1 = 0.f, px2 = 0.f, py2 = 0.f;
    if (vb) {
        const float* pb = pd_bboxes + ba * 4;
        px1 = pb[0]; py1 = pb[1]; px2 = pb[2]; py2 = pb[3];
        const float* srow = pd_scores + ba * NCLS;
        unsigned long long bits = vb;
        while (bits) {
            int n = __ffsll(bits) - 1; bits &= bits - 1;
            float gx1 = gtb[n * 4 + 0], gy1 = gtb[n * 4 + 1];
            float gx2 = gtb[n * 4 + 2], gy2 = gtb[n * 4 + 3];
            float garea = (gx2 - gx1) * (gy2 - gy1);
            float m = metric_valid(gx1, gy1, gx2, gy2, garea,
                                   px1, py1, px2, py2, srow[lab[n]]);
            maxAll = fmaxf(maxAll, m);
            if ((pos >> n) & 1ull) maxPos = fmaxf(maxPos, m);
        }
    }

    int fg = __popcll(pos);
    float pal;
    if (fg > 1) {
        // conflict: keep only gts whose metric equals the column max
        unsigned long long eq = 0ull, bits = vb;
        const float* srow = pd_scores + ba * NCLS;
        while (bits) {
            int n = __ffsll(bits) - 1; bits &= bits - 1;
            float gx1 = gtb[n * 4 + 0], gy1 = gtb[n * 4 + 1];
            float gx2 = gtb[n * 4 + 2], gy2 = gtb[n * 4 + 3];
            float garea = (gx2 - gx1) * (gy2 - gy1);
            float m = metric_valid(gx1, gy1, gx2, gy2, garea,
                                   px1, py1, px2, py2, srow[lab[n]]);
            if (m == maxAll) eq |= 1ull << n;
        }
        pos &= eq;
        fg = __popcll(pos);
        pal = pos ? maxAll : 0.f;
    } else {
        pal = fg ? maxPos : 0.f;
    }
    const int idx = pos ? (__ffsll(pos) - 1) : 0;   // argmax of 0/1 over n

    out_fg[ba] = (fg > 0) ? 1.f : 0.f;
    out_bboxes[ba * 4 + 0] = gtb[idx * 4 + 0];
    out_bboxes[ba * 4 + 1] = gtb[idx * 4 + 1];
    out_bboxes[ba * 4 + 2] = gtb[idx * 4 + 2];
    out_bboxes[ba * 4 + 3] = gtb[idx * 4 + 3];
    tgt_idx[ba] = idx;
    pos_align_out[ba] = pal;
}

// ---------------------------------------------------------------------------
// Kernel C: target_scores via WMMA. One wave per 16-anchor tile.
//   D(16 anchors x 80 classes) = E(16x64 one-hot of target_gt_idx)
//                              x L(64x80 one-hot of gt_labels)
// Operands are exact 0/1 f16; f32 accumulation => exact one-hot result,
// scaled by pos_align in f32. Two K-chained v_wmma_f32_16x16x32_f16 per
// 16-class tile, 5 tiles. VGPR layouts per ISA 7.12.2.
// ---------------------------------------------------------------------------
__global__ __launch_bounds__(32) void scores_wmma_kernel(
    const int* __restrict__ gt_labels, const int* __restrict__ tgt_idx,
    const float* __restrict__ pos_align, float* __restrict__ out_scores)
{
    const int tile = blockIdx.x % (NA / 16);
    const int b    = blockIdx.x / (NA / 16);
    const int a0   = tile * 16;
    const int l    = threadIdx.x;
    const int col  = l & 15;
    const bool hi  = l >= 16;

    // A fragment: lane l holds row M = l%16; element (2v+s) covers
    // K = [(v<4 ? 2v+s : 16+2(v-4)+s)] + (hi ? 8 : 0)
    const int idxM = tgt_idx[(size_t)b * NA + a0 + col];
    v16h A0, A1;
    #pragma unroll
    for (int v = 0; v < 8; ++v) {
        #pragma unroll
        for (int s = 0; s < 2; ++s) {
            int base = (v < 4) ? (2 * v + s) : (16 + 2 * (v - 4) + s);
            int K = base + (hi ? 8 : 0);
            A0[2 * v + s] = (idxM == K)      ? (_Float16)1.0f : (_Float16)0.0f;
            A1[2 * v + s] = (idxM == K + 32) ? (_Float16)1.0f : (_Float16)0.0f;
        }
    }

    // B fragment: lane l holds column N = l%16; element e covers
    // K = (e < 8 ? e : e + 8) + (hi ? 8 : 0)  (same K order as A)
    int labK[16], labK2[16];
    #pragma unroll
    for (int e = 0; e < 16; ++e) {
        int Kb = (e < 8 ? e : e + 8) + (hi ? 8 : 0);
        labK[e]  = gt_labels[b * NGT + Kb];
        labK2[e] = gt_labels[b * NGT + Kb + 32];
    }

    float pal[8];
    #pragma unroll
    for (int r = 0; r < 8; ++r)
        pal[r] = pos_align[(size_t)b * NA + a0 + r + (hi ? 8 : 0)];

    for (int ct = 0; ct < 5; ++ct) {
        const int c = ct * 16 + col;
        v16h B0, B1;
        #pragma unroll
        for (int e = 0; e < 16; ++e) {
            B0[e] = (labK[e]  == c) ? (_Float16)1.0f : (_Float16)0.0f;
            B1[e] = (labK2[e] == c) ? (_Float16)1.0f : (_Float16)0.0f;
        }
        v8f acc = {};
        acc = __builtin_amdgcn_wmma_f32_16x16x32_f16(false, A0, false, B0,
                                                     (short)0, acc, false, false);
        acc = __builtin_amdgcn_wmma_f32_16x16x32_f16(false, A1, false, B1,
                                                     (short)0, acc, false, false);
        #pragma unroll
        for (int r = 0; r < 8; ++r) {
            int m = r + (hi ? 8 : 0);   // D: VGPR r -> row M (+8 for hi lanes)
            out_scores[((size_t)b * NA + a0 + m) * NCLS + c] = acc[r] * pal[r];
        }
    }
}

extern "C" void kernel_launch(void* const* d_in, const int* in_sizes, int n_in,
                              void* d_out, int out_size, void* d_ws, size_t ws_size,
                              hipStream_t stream)
{
    const float* pd_scores = (const float*)d_in[0];   // (32, 8400, 80)
    const float* pd_bboxes = (const float*)d_in[1];   // (32, 8400, 4)
    const float* anc       = (const float*)d_in[2];   // (8400, 2)
    const int*   gt_labels = (const int*)d_in[3];     // (32, 64, 1)
    const float* gt_bboxes = (const float*)d_in[4];   // (32, 64, 4)
    const float* mask_gt   = (const float*)d_in[5];   // (32, 64, 1)

    const size_t BA = (size_t)NB * NA;
    unsigned long long* valid_bits = (unsigned long long*)d_ws;
    unsigned long long* topk_bits  = valid_bits + BA;
    int*   tgt_idx = (int*)(topk_bits + BA);
    float* pal     = (float*)(tgt_idx + BA);

    float* out        = (float*)d_out;
    float* out_bboxes = out;                      // (32, 8400, 4)
    float* out_scores = out + BA * 4;             // (32, 8400, 80)
    float* out_fg     = out_scores + BA * NCLS;   // (32, 8400)

    // zero the bitmask scratch (deterministic per call)
    hipMemsetAsync(d_ws, 0, BA * 16, stream);

    topk_kernel<<<dim3(NB * NGT), 256, 0, stream>>>(
        pd_scores, pd_bboxes, anc, gt_labels, gt_bboxes, mask_gt,
        valid_bits, topk_bits);

    assign_kernel<<<dim3((NA + 255) / 256, NB), 256, 0, stream>>>(
        pd_scores, pd_bboxes, gt_labels, gt_bboxes,
        valid_bits, topk_bits, out_bboxes, out_fg, tgt_idx, pal);

    scores_wmma_kernel<<<dim3(NB * (NA / 16)), 32, 0, stream>>>(
        gt_labels, tgt_idx, pal, out_scores);
}
